// ReMambaBlock_40346922778718
// MI455X (gfx1250) — matrix operand order repaired and checked by reading, not compiled, not run on hardware
//
#include <hip/hip_runtime.h>
#include <hip/hip_bf16.h>

// ReMambaBlock for gfx1250 (MI455X): bf16 WMMA GEMMs + register-resident scan.
// Shapes: B=2 L=8 C=128 H=W=32; d_inner=256 d_state=16 dt_rank=8 d_conv=4.

typedef __attribute__((ext_vector_type(16))) __bf16 v16bf;
typedef __attribute__((ext_vector_type(8)))  __bf16 v8bf;
typedef __attribute__((ext_vector_type(8)))  float  v8f;

__device__ __forceinline__ float siluf(float x)     { return x / (1.0f + __expf(-x)); }
__device__ __forceinline__ float softplusf(float x) { return (x > 20.0f) ? x : log1pf(__expf(x)); }

__device__ __forceinline__ unsigned pack_bf16(float a, float b) {
    union { __bf16 h[2]; unsigned u; } p;
    p.h[0] = (__bf16)a; p.h[1] = (__bf16)b;
    return p.u;
}

// ---------------------------------------------------------------------------
// Prep kernels
// ---------------------------------------------------------------------------

// Wdelta[256][256] = x_proj_w[:, 0:8] @ dt_proj_w[8][256]   (fold rank-8 proj)
__global__ __launch_bounds__(256) void prep_wdelta(const float* __restrict__ xpw,
                                                   const float* __restrict__ dtw,
                                                   float* __restrict__ wd) {
    int i = blockIdx.x;        // 0..255 (d_inner row)
    int j = threadIdx.x;       // 0..255 (d_inner col)
    float s = 0.f;
#pragma unroll
    for (int r = 0; r < 8; ++r) s += xpw[i * 40 + r] * dtw[r * 256 + j];
    wd[i * 256 + j] = s;
}

// Aneg = -exp(A_log), 256*16 elements
__global__ __launch_bounds__(256) void prep_negexpA(const float* __restrict__ Alog,
                                                    float* __restrict__ Aneg) {
    int i = blockIdx.x * 256 + threadIdx.x;
    Aneg[i] = -__expf(Alog[i]);
}

// x_c input: mean over H*W of x plus prev_x.  grid = B*L*C = 2048 blocks.
__global__ __launch_bounds__(256) void mean_kernel(const float* __restrict__ x,
                                                   const float* __restrict__ prev,
                                                   float* __restrict__ xin) {
    __shared__ float red[256];
    int blc = blockIdx.x;                       // flat (b,l,c)
    const float* p = x + (long)blc * 1024;      // H*W contiguous
    float s = 0.f;
    for (int i = threadIdx.x; i < 1024; i += 256) s += p[i];
    red[threadIdx.x] = s;
    __syncthreads();
    for (int off = 128; off > 0; off >>= 1) {
        if (threadIdx.x < off) red[threadIdx.x] += red[threadIdx.x + off];
        __syncthreads();
    }
    if (threadIdx.x == 0) xin[blc] = red[0] * (1.0f / 1024.0f) + prev[blc];
}

// ---------------------------------------------------------------------------
// Depthwise causal conv (width 4) + SiLU.  grid = nbatch*seqlen, block = 256(d)
// ---------------------------------------------------------------------------
__global__ __launch_bounds__(256) void conv_silu(const float* __restrict__ xm_pre,
                                                 const float* __restrict__ cw,
                                                 const float* __restrict__ cb,
                                                 float* __restrict__ xm,
                                                 int seqlen) {
    int bl = blockIdx.x / seqlen;
    int t  = blockIdx.x % seqlen;
    int d  = threadIdx.x;
    const float* base = xm_pre + (long)bl * seqlen * 256 + d;
    float s = cb[d];
#pragma unroll
    for (int k = 0; k < 4; ++k) {
        int tt = t - 3 + k;
        if (tt >= 0) s += base[(long)tt * 256] * cw[d * 4 + k];
    }
    xm[((long)bl * seqlen + t) * 256 + d] = siluf(s);
}

// ---------------------------------------------------------------------------
// Selective scan: one thread per (batch, d_inner channel), 16 states in VGPRs.
// grid = nbatch, block = 256
// ---------------------------------------------------------------------------
__global__ __launch_bounds__(256) void scan_kernel(const float* __restrict__ u,
                                                   const float* __restrict__ delta,
                                                   const float* __restrict__ BC,
                                                   const float* __restrict__ Aneg,
                                                   const float* __restrict__ Dv,
                                                   float* __restrict__ y,
                                                   int seqlen) {
    __shared__ float sbc[32];
    int bl = blockIdx.x;
    int d  = threadIdx.x;
    float a[16], h[16];
#pragma unroll
    for (int n = 0; n < 16; ++n) { a[n] = Aneg[d * 16 + n]; h[n] = 0.f; }
    float Dd = Dv[d];
    long base = (long)bl * seqlen;
    for (int t = 0; t < seqlen; ++t) {
        long row = base + t;
        if (threadIdx.x < 32) sbc[threadIdx.x] = BC[row * 32 + threadIdx.x];
        __syncthreads();
        float dl = delta[row * 256 + d];
        float uu = u[row * 256 + d];
        float du = dl * uu;
        float acc = 0.f;
#pragma unroll
        for (int n = 0; n < 16; ++n) {
            float dA = __expf(dl * a[n]);       // v_exp_f32 (trans, co-issues)
            h[n] = dA * h[n] + du * sbc[n];     // B at sbc[0..15]
            acc += h[n] * sbc[16 + n];          // C at sbc[16..31]
        }
        y[row * 256 + d] = acc + uu * Dd;
        __syncthreads();
    }
}

// ---------------------------------------------------------------------------
// Generic bf16 WMMA GEMM: Out = epi(A[M,K] @ W[K,N]).
// One wave per 16(M)x32(N) macro-tile (two WMMA accumulators sharing one A
// fragment), 8 waves/block, per-wave LDS staging (intra-wave DS ordering, no
// barriers).  B staged transposed [n][k] so fragments are contiguous b128 runs.
// AMODE: 0 = row-major A (lda=K)
//        1 = gather from x: A[m,k] = x[((m/1024)*K + k)*1024 + m%1024]
//        2 = A[m,k] = Y[m*K+k] * silu(Z[m*K+k])
// EPI:   0 = plain store  out[m*Ncols + n]
//        1 = split store: n<256 -> out, else -> out2 (in_proj xm/z split)
//        2 = softplus(acc + bias[n]) -> out[m*Ncols+n]   (bias via `resid`)
//        3 = scatter+residual: out[((m/1024)*Ncols+n)*1024 + m%1024] = acc+resid
// ---------------------------------------------------------------------------
template <int AMODE, int EPI>
__global__ __launch_bounds__(256) void gemm16(const float* __restrict__ A,
                                              const float* __restrict__ Z,
                                              const float* __restrict__ W,
                                              int ldw, int wcol0,
                                              float* __restrict__ out,
                                              float* __restrict__ out2,
                                              const float* __restrict__ resid,
                                              int M, int Ncols, int K) {
    __shared__ __bf16 sA[8][16 * 32];   // [m][k]
    __shared__ __bf16 sB[8][32 * 32];   // transposed: [n][k]
    const int wave = threadIdx.x >> 5;
    const int lane = threadIdx.x & 31;
    const long Mt = M >> 4, Nt2 = Ncols >> 5;
    long tile = (long)blockIdx.x * 8 + wave;
    if (tile >= Mt * Nt2) return;
    const int tn = (int)(tile % Nt2);
    const int tm = (int)(tile / Nt2);
    const int m0 = tm << 4, n0 = tn << 5;

    __bf16* la = sA[wave];
    __bf16* lb = sB[wave];
    unsigned* laU = (unsigned*)la;
    unsigned* lbU = (unsigned*)lb;
    v8f acc0 = {}, acc1 = {};

    const int mrow = lane & 15;
    const int half = lane >> 4;

    for (int k0 = 0; k0 < K; k0 += 32) {
        // stage A tile 16x32: packed bf16 pair per store (ds_store_b32)
        for (int i = lane; i < 256; i += 32) {
            int r = i >> 4, kp = (i & 15) << 1;     // row, even k
            int m = m0 + r, k = k0 + kp;
            float v0, v1;
            if (AMODE == 0) {
                long idx = (long)m * K + k;
                v0 = A[idx]; v1 = A[idx + 1];
            } else if (AMODE == 1) {
                long bidx = ((long)(m >> 10) * K + k) * 1024 + (m & 1023);
                v0 = A[bidx]; v1 = A[bidx + 1024];
            } else {
                long idx = (long)m * K + k;
                v0 = A[idx] * siluf(Z[idx]);
                v1 = A[idx + 1] * siluf(Z[idx + 1]);
            }
            laU[(r << 4) + (kp >> 1)] = pack_bf16(v0, v1);
        }
        // stage B tile 32(k)x32(n), transposed into LDS as [n][k]
        for (int i = lane; i < 512; i += 32) {
            int nn = i & 31, kp = (i >> 5) << 1;    // col, even k
            long widx = (long)(k0 + kp) * ldw + wcol0 + n0 + nn;
            lbU[(nn << 4) + (kp >> 1)] = pack_bf16(W[widx], W[widx + ldw]);
        }
        // fragments: A lane run = one contiguous 16-elt group (k = 16*half + s),
        // B lane run = one contiguous 16-elt group in transposed LDS.
        const v8bf* laV = (const v8bf*)la;
        const v8bf* lbV = (const v8bf*)lb;
        v8bf a0 = laV[(mrow << 2) + (half << 1)];      // la[m*32 + half*16 .. +7]
        v8bf a1 = laV[(mrow << 2) + (half << 1) + 1];  // la[m*32 + half*16 + 8 ..]
        v16bf af = __builtin_shufflevector(a0, a1, 0,1,2,3,4,5,6,7,8,9,10,11,12,13,14,15);
        int bbase0 = (mrow << 2) + (half << 1);            // (n=mrow)*32 + half*16
        int bbase1 = ((16 + mrow) << 2) + (half << 1);     // (n=16+mrow)*32 + ...
        v8bf b0l = lbV[bbase0],     b0h = lbV[bbase0 + 1];
        v8bf b1l = lbV[bbase1],     b1h = lbV[bbase1 + 1];
        v16bf bf0 = __builtin_shufflevector(b0l, b0h, 0,1,2,3,4,5,6,7,8,9,10,11,12,13,14,15);
        v16bf bf1 = __builtin_shufflevector(b1l, b1h, 0,1,2,3,4,5,6,7,8,9,10,11,12,13,14,15);
        acc0 = __builtin_amdgcn_wmma_f32_16x16x32_bf16(false, af, false, bf0, (short)0, acc0, false, false);
        acc1 = __builtin_amdgcn_wmma_f32_16x16x32_bf16(false, af, false, bf1, (short)0, acc1, false, false);
    }

    // epilogue: C/D layout -> lane holds column n, VGPR v -> row v (+8 hi half)
#pragma unroll
    for (int j = 0; j < 2; ++j) {
        const v8f& acc = j ? acc1 : acc0;
        const int nj = n0 + (j << 4) + mrow;        // global col; (n0+j*16) uniform
#pragma unroll
        for (int v = 0; v < 8; ++v) {
            int m = m0 + v + (half << 3);
            float r = acc[v];
            if (EPI == 0) {
                out[(long)m * Ncols + nj] = r;
            } else if (EPI == 1) {
                // uniform branch: whole 16-wide tile is on one side of 256
                if (n0 + (j << 4) < 256) out[(long)m * 256 + nj] = r;
                else                     out2[(long)m * 256 + (nj - 256)] = r;
            } else if (EPI == 2) {
                out[(long)m * Ncols + nj] = softplusf(r + resid[nj]);
            } else {
                long bl = m >> 10, hw = m & 1023;
                long idx = (bl * Ncols + nj) * 1024 + hw;
                out[idx] = r + resid[idx];
            }
        }
    }
}

// ---------------------------------------------------------------------------
extern "C" void kernel_launch(void* const* d_in, const int* in_sizes, int n_in,
                              void* d_out, int out_size, void* d_ws, size_t ws_size,
                              hipStream_t stream) {
    const float* x      = (const float*)d_in[0];
    const float* prev_x = (const float*)d_in[1];
    const float* f_inw  = (const float*)d_in[2];
    const float* f_cw   = (const float*)d_in[3];
    const float* f_cb   = (const float*)d_in[4];
    const float* f_xpw  = (const float*)d_in[5];
    const float* f_dtw  = (const float*)d_in[6];
    const float* f_dtb  = (const float*)d_in[7];
    const float* f_Alog = (const float*)d_in[8];
    const float* f_D    = (const float*)d_in[9];
    const float* f_outw = (const float*)d_in[10];
    const float* c_inw  = (const float*)d_in[11];
    const float* c_cw   = (const float*)d_in[12];
    const float* c_cb   = (const float*)d_in[13];
    const float* c_xpw  = (const float*)d_in[14];
    const float* c_dtw  = (const float*)d_in[15];
    const float* c_dtb  = (const float*)d_in[16];
    const float* c_Alog = (const float*)d_in[17];
    const float* c_D    = (const float*)d_in[18];
    const float* c_outw = (const float*)d_in[19];

    const int FM = 16384;   // f-branch tokens (B*L * H*W)
    const int CM = 16;      // c-branch tokens (B*L)

    float* ws = (float*)d_ws;
    size_t o = 0;
    auto alloc = [&](size_t n) { float* p = ws + o; o += n; return p; };
    float* f_xmp = alloc((size_t)FM * 256);   // in_proj xm (pre-conv); reused as delta
    float* f_z   = alloc((size_t)FM * 256);
    float* f_xm  = alloc((size_t)FM * 256);
    float* f_y   = alloc((size_t)FM * 256);
    float* f_BC  = alloc((size_t)FM * 32);
    float* f_delta = f_xmp;                   // reuse: xm_pre dead after conv
    float* wd_f  = alloc(256 * 256);
    float* wd_c  = alloc(256 * 256);
    float* An_f  = alloc(256 * 16);
    float* An_c  = alloc(256 * 16);
    float* c_xin = alloc(16 * 128);
    float* c_xmp = alloc(16 * 256);
    float* c_z   = alloc(16 * 256);
    float* c_xm  = alloc(16 * 256);
    float* c_y   = alloc(16 * 256);
    float* c_BC  = alloc(16 * 32);
    float* c_delta = c_xmp;

    float* out_f = (float*)d_out;             // (2,8,128,32,32)
    float* out_c = (float*)d_out + 2097152;   // (2,8,128)

    auto blocks = [](long tiles) { return (unsigned)((tiles + 7) / 8); };

    // ---- prep ----
    prep_wdelta<<<256, 256, 0, stream>>>(f_xpw, f_dtw, wd_f);
    prep_wdelta<<<256, 256, 0, stream>>>(c_xpw, c_dtw, wd_c);
    prep_negexpA<<<16, 256, 0, stream>>>(f_Alog, An_f);
    prep_negexpA<<<16, 256, 0, stream>>>(c_Alog, An_c);
    mean_kernel<<<2048, 256, 0, stream>>>(x, prev_x, c_xin);

    // ---- f branch ----  (wave macro-tile = 16x32 -> tiles = (M/16)*(N/32))
    gemm16<1, 1><<<blocks(1024L * 16), 256, 0, stream>>>(
        x, nullptr, f_inw, 512, 0, f_xmp, f_z, nullptr, FM, 512, 128);
    conv_silu<<<16 * 1024, 256, 0, stream>>>(f_xmp, f_cw, f_cb, f_xm, 1024);
    gemm16<0, 2><<<blocks(1024L * 8), 256, 0, stream>>>(
        f_xm, nullptr, wd_f, 256, 0, f_delta, nullptr, f_dtb, FM, 256, 256);
    gemm16<0, 0><<<blocks(1024L * 1), 256, 0, stream>>>(
        f_xm, nullptr, f_xpw, 40, 8, f_BC, nullptr, nullptr, FM, 32, 256);
    scan_kernel<<<16, 256, 0, stream>>>(f_xm, f_delta, f_BC, An_f, f_D, f_y, 1024);
    gemm16<2, 3><<<blocks(1024L * 4), 256, 0, stream>>>(
        f_y, f_z, f_outw, 128, 0, out_f, nullptr, x, FM, 128, 256);

    // ---- c branch ----
    gemm16<0, 1><<<blocks(1L * 16), 256, 0, stream>>>(
        c_xin, nullptr, c_inw, 512, 0, c_xmp, c_z, nullptr, CM, 512, 128);
    conv_silu<<<16, 256, 0, stream>>>(c_xmp, c_cw, c_cb, c_xm, 8);
    gemm16<0, 2><<<blocks(1L * 8), 256, 0, stream>>>(
        c_xm, nullptr, wd_c, 256, 0, c_delta, nullptr, c_dtb, CM, 256, 256);
    gemm16<0, 0><<<blocks(1L * 1), 256, 0, stream>>>(
        c_xm, nullptr, c_xpw, 40, 8, c_BC, nullptr, nullptr, CM, 32, 256);
    scan_kernel<<<2, 256, 0, stream>>>(c_xm, c_delta, c_BC, An_c, c_D, c_y, 8);
    gemm16<2, 0><<<blocks(1L * 4), 256, 0, stream>>>(
        c_y, c_z, c_outw, 128, 0, out_c, nullptr, nullptr, CM, 128, 256);
}
